// FusionRestormer_83288005804523
// MI455X (gfx1250) — matrix operand order repaired and checked by reading, not compiled, not run on hardware
//
#include <hip/hip_runtime.h>
#include <hip/hip_bf16.h>
#include <math.h>

// ---------------------------------------------------------------------------
// FusionRestormer on MI455X (gfx1250).
//  - All matmuls on v_wmma_f32_16x16x32_f16, register-blocked 4x per wave.
//  - GEMM inputs kept as channel-last f16 mirrors (pre-padded, pre-activated)
//    so LDS staging is a contiguous DMA: global_load_async_to_lds_b128
//    (ASYNCcnt + s_wait_asynccnt), no VALU staging.
//  - Attention (norms / q.kT / attn.v) reads f16 NCHW mirror directly.
// ---------------------------------------------------------------------------

typedef __attribute__((ext_vector_type(16))) _Float16 v16h;
typedef __attribute__((ext_vector_type(8)))  _Float16 v8h;
typedef __attribute__((ext_vector_type(8)))  float    v8f;

#define HW_   25600      // 160*160
#define WID_  160
#define DCH   96
#define FFH_  255
#define LBLK  4
#define HEADS 2

__device__ __forceinline__ float gelu_f(float x) {
  return 0.5f * x * (1.0f + erff(x * 0.70710678118654752440f));
}

__device__ __forceinline__ v8f wmma16(v16h a, v16h b, v8f c) {
  return __builtin_amdgcn_wmma_f32_16x16x32_f16(false, a, false, b, (short)0, c,
                                                false, false);
}

// A fragment (16x32 f16, MxK), source row-major [M][ldk] halves.
__device__ __forceinline__ v16h frag_a_f16(const _Float16* base, int ldk, int kb) {
  int lane = threadIdx.x & 31;
  int m    = lane & 15;
  int koff = (lane < 16) ? 0 : 8;
  const _Float16* p = base + (size_t)m * ldk + kb + koff;
  v8h lo = *(const v8h*)(p);
  v8h hi = *(const v8h*)(p + 16);
  return __builtin_shufflevector(lo, hi, 0, 1, 2, 3, 4, 5, 6, 7,
                                 8, 9, 10, 11, 12, 13, 14, 15);
}

// B fragment (32x16 f16, KxN), source B^T row-major [N][ldk] halves.
__device__ __forceinline__ v16h frag_b_f16(const _Float16* base, int ldk, int kb) {
  int lane = threadIdx.x & 31;
  int n    = lane & 15;
  int koff = (lane < 16) ? 0 : 8;
  const _Float16* p = base + (size_t)n * ldk + kb + koff;
  v8h lo = *(const v8h*)(p);
  v8h hi = *(const v8h*)(p + 16);
  return __builtin_shufflevector(lo, hi, 0, 1, 2, 3, 4, 5, 6, 7,
                                 8, 9, 10, 11, 12, 13, 14, 15);
}

// Async global->LDS DMA (CDNA5). 16 bytes per lane per issue; ASYNCcnt.
__device__ __forceinline__ void async_copy_b128(unsigned ldsOff, const void* g) {
  asm volatile("global_load_async_to_lds_b128 %0, %1, off"
               :: "v"(ldsOff), "v"((unsigned long long)g) : "memory");
}
__device__ __forceinline__ void wait_async0() {
  asm volatile("s_wait_asynccnt 0x0" ::: "memory");
}

// ---------------------------------------------------------------------------
// Weight packing: f32 (Cout,Cin[,3,3]) -> f16 [NPAD][KPAD], zero padded.
// ---------------------------------------------------------------------------
__global__ void pack_w_1x1(const float* __restrict__ W, _Float16* __restrict__ P,
                           int Cout, int Cin, int KPAD, int NPAD) {
  int i = blockIdx.x * blockDim.x + threadIdx.x;
  if (i >= NPAD * KPAD) return;
  int n = i / KPAD, k = i - n * KPAD;
  float v = (n < Cout && k < Cin) ? W[(size_t)n * Cin + k] : 0.0f;
  P[i] = (_Float16)v;
}

__global__ void pack_w_3x3(const float* __restrict__ W, _Float16* __restrict__ P,
                           int Cout, int Cin, int NPAD) {
  int KP = Cin * 9;
  int i  = blockIdx.x * blockDim.x + threadIdx.x;
  if (i >= NPAD * KP) return;
  int n = i / KP, k = i - n * KP;
  float v = (n < Cout) ? W[(size_t)n * KP + k] : 0.0f;
  P[i] = (_Float16)v;
}

// ---------------------------------------------------------------------------
// 1x1-conv GEMM. Input: f16 channel-last mirror XH [img*HW][KPAD] (padded,
// pre-activated). WG = 64 pixels x 64 couts, 4 waves, 4 acc/wave.
// grid = (NIMG*HW/64, ceil(Cout/64)); dyn-LDS = 64*KPAD*2 bytes.
// Epilogue: optional f32 out (+bias,+resid) and/or f16 channel-last mirror
// (with optional GELU) for the next consumer.
// ---------------------------------------------------------------------------
__global__ void gemm1x1_wmma(const _Float16* __restrict__ XH,
                             const _Float16* __restrict__ Wp,
                             const float* __restrict__ bias,
                             const float* __restrict__ resid,
                             float* __restrict__ Yf,
                             _Float16* __restrict__ Ymir,
                             int KPAD, int Cout, int geluMir) {
  extern __shared__ _Float16 lds[];  // [64][KPAD]
  int gp0 = blockIdx.x * 64;
  int img = gp0 / HW_;
  int p0  = gp0 - img * HW_;

  // contiguous DMA: 64 rows * KPAD halves
  unsigned ldsBase = (unsigned)(unsigned long long)(void*)lds;
  const char* gsrc = (const char*)(XH + (size_t)gp0 * KPAD);
  int nchunk = 8 * KPAD;  // 16B chunks; multiple of 128
  for (int i = threadIdx.x; i < nchunk; i += 128)
    async_copy_b128(ldsBase + i * 16, gsrc + (size_t)i * 16);
  wait_async0();
  __syncthreads();

  int wave     = threadIdx.x >> 5;
  int coutBase = blockIdx.y * 64 + wave * 16;
  const _Float16* Arow = Wp + (size_t)coutBase * KPAD;

  v8f acc0 = {}, acc1 = {}, acc2 = {}, acc3 = {};
  for (int kb = 0; kb < KPAD; kb += 32) {
    v16h a  = frag_a_f16(Arow, KPAD, kb);
    v16h b0 = frag_b_f16(lds,               KPAD, kb);
    v16h b1 = frag_b_f16(lds + 16 * KPAD,   KPAD, kb);
    v16h b2 = frag_b_f16(lds + 32 * KPAD,   KPAD, kb);
    v16h b3 = frag_b_f16(lds + 48 * KPAD,   KPAD, kb);
    acc0 = wmma16(a, b0, acc0);
    acc1 = wmma16(a, b1, acc1);
    acc2 = wmma16(a, b2, acc2);
    acc3 = wmma16(a, b3, acc3);
  }

  int lane = threadIdx.x & 31;
  int n    = lane & 15;
  int moff = (lane < 16) ? 0 : 8;
  v8f* accs[4] = {&acc0, &acc1, &acc2, &acc3};
#pragma unroll
  for (int sub = 0; sub < 4; ++sub) {
    v8f av = *accs[sub];
    int pix = p0 + sub * 16 + n;
    if (Yf) {
      size_t obase = (size_t)img * Cout * HW_ + pix;
#pragma unroll
      for (int r = 0; r < 8; ++r) {
        int co = coutBase + moff + r;
        if (co < Cout) {
          size_t o = obase + (size_t)co * HW_;
          float v = av[r] + bias[co];
          if (resid) v += resid[o];
          Yf[o] = v;
        }
      }
    }
    if (Ymir) {  // used only for Cout==96 paths; full 8-wide stores
      v8h o;
#pragma unroll
      for (int r = 0; r < 8; ++r) {
        float v = av[r] + bias[coutBase + moff + r];
        if (geluMir) v = gelu_f(v);
        o[r] = (_Float16)v;
      }
      *(v8h*)(Ymir + ((size_t)img * HW_ + pix) * DCH + coutBase + moff) = o;
    }
  }
}

// ---------------------------------------------------------------------------
// Dense 3x3 conv as implicit GEMM: K = 96*9 = 864, pad=1. Input is f16
// channel-last mirror [img*HW][96] (pre-activated). WG = 32 px x 64 couts.
// ---------------------------------------------------------------------------
__global__ void conv3x3_wmma(const _Float16* __restrict__ XH,
                             const _Float16* __restrict__ Wp,
                             const float* __restrict__ bias,
                             float* __restrict__ Y, int Cout) {
  const int KP = 864;
  extern __shared__ _Float16 lds[];  // [32][864]
  int gp0 = blockIdx.x * 32;
  int img = gp0 / HW_;
  int p0  = gp0 - img * HW_;
  const _Float16* Xi = XH + (size_t)img * HW_ * DCH;

  for (int e = threadIdx.x; e < 32 * KP; e += blockDim.x) {
    int pix = e & 31, k = e >> 5;
    int ci = k / 9, rr = k - ci * 9, ky = rr / 3, kx = rr - ky * 3;
    int p = p0 + pix, y = p / WID_, x = p - y * WID_;
    int ny = y + ky - 1, nx = x + kx - 1;
    _Float16 v = (_Float16)0.0f;
    if ((unsigned)ny < 160u && (unsigned)nx < 160u)
      v = Xi[(size_t)(ny * WID_ + nx) * DCH + ci];
    lds[pix * KP + k] = v;
  }
  __syncthreads();

  int wave     = threadIdx.x >> 5;
  int coutBase = blockIdx.y * 64 + wave * 16;
  const _Float16* Arow = Wp + (size_t)coutBase * KP;

  v8f acc0 = {}, acc1 = {};
  for (int kb = 0; kb < KP; kb += 32) {
    v16h a  = frag_a_f16(Arow, KP, kb);
    v16h b0 = frag_b_f16(lds,            KP, kb);
    v16h b1 = frag_b_f16(lds + 16 * KP,  KP, kb);
    acc0 = wmma16(a, b0, acc0);
    acc1 = wmma16(a, b1, acc1);
  }

  int lane = threadIdx.x & 31;
  int n    = lane & 15;
  int moff = (lane < 16) ? 0 : 8;
#pragma unroll
  for (int sub = 0; sub < 2; ++sub) {
    v8f av = sub ? acc1 : acc0;
    size_t obase = (size_t)img * Cout * HW_ + p0 + sub * 16 + n;
#pragma unroll
    for (int r = 0; r < 8; ++r) {
      int co = coutBase + moff + r;
      if (co < Cout) Y[obase + (size_t)co * HW_] = av[r] + bias[co];
    }
  }
}

// ---------------------------------------------------------------------------
// Stem conv1: 3->96, 3x3, pad1. Writes GELU'd f16 channel-last mirror only.
// ---------------------------------------------------------------------------
__global__ void stem1_k(const float* __restrict__ X, const float* __restrict__ W,
                        const float* __restrict__ b, _Float16* __restrict__ YH) {
  int idx = blockIdx.x * blockDim.x + threadIdx.x;  // 6*96*HW
  if (idx >= 6 * DCH * HW_) return;
  int pix = idx % HW_;
  int t   = idx / HW_;
  int co  = t % DCH;
  int img = t / DCH;
  int y = pix / WID_, x = pix - y * WID_;
  const float* Xi = X + (size_t)img * 3 * HW_;
  float acc = b[co];
#pragma unroll
  for (int ci = 0; ci < 3; ++ci)
#pragma unroll
    for (int ky = 0; ky < 3; ++ky) {
      int ny = y + ky - 1;
      if ((unsigned)ny >= 160u) continue;
#pragma unroll
      for (int kx = 0; kx < 3; ++kx) {
        int nx = x + kx - 1;
        if ((unsigned)nx >= 160u) continue;
        acc += W[(co * 3 + ci) * 9 + ky * 3 + kx] * Xi[(size_t)ci * HW_ + ny * WID_ + nx];
      }
    }
  YH[((size_t)img * HW_ + pix) * DCH + co] = (_Float16)gelu_f(acc);
}

// Depthwise 3x3 + bias; writes f32 or f16 (NCHW) depending on non-null out.
__global__ void dwconv3x3_k(const float* __restrict__ X, const float* __restrict__ Wd,
                            const float* __restrict__ b, float* __restrict__ Yf,
                            _Float16* __restrict__ Yh, int C, int NIMG) {
  long long idx = (long long)blockIdx.x * blockDim.x + threadIdx.x;
  long long tot = (long long)NIMG * C * HW_;
  if (idx >= tot) return;
  int pix = (int)(idx % HW_);
  long long t = idx / HW_;
  int c = (int)(t % C);
  int y = pix / WID_, x = pix - y * WID_;
  const float* Xi = X + (size_t)t * HW_;
  float acc = b[c];
#pragma unroll
  for (int ky = 0; ky < 3; ++ky) {
    int ny = y + ky - 1;
    if ((unsigned)ny >= 160u) continue;
#pragma unroll
    for (int kx = 0; kx < 3; ++kx) {
      int nx = x + kx - 1;
      if ((unsigned)nx >= 160u) continue;
      acc += Wd[c * 9 + ky * 3 + kx] * Xi[ny * WID_ + nx];
    }
  }
  if (Yf) Yf[idx] = acc;
  else    Yh[idx] = (_Float16)acc;
}

// LayerNorm2d over 96 channels per pixel; writes f16 channel-last mirror.
__global__ void ln2d_k(const float* __restrict__ X, const float* __restrict__ w,
                       const float* __restrict__ b, _Float16* __restrict__ YH,
                       int NIMG) {
  int gp = blockIdx.x * blockDim.x + threadIdx.x;
  if (gp >= NIMG * HW_) return;
  int img = gp / HW_, p = gp - img * HW_;
  const float* Xi = X + (size_t)img * DCH * HW_ + p;
  float s = 0.f, ss = 0.f;
#pragma unroll 4
  for (int c = 0; c < DCH; ++c) {
    float v = Xi[(size_t)c * HW_];
    s += v; ss += v * v;
  }
  float m   = s * (1.0f / DCH);
  float var = fmaxf(ss * (1.0f / DCH) - m * m, 0.0f);
  float inv = rsqrtf(var + 1e-6f);
  _Float16* Yi = YH + (size_t)gp * DCH;
#pragma unroll 4
  for (int c = 0; c < DCH; ++c)
    Yi[c] = (_Float16)((Xi[(size_t)c * HW_] - m) * inv * w[c] + b[c]);
}

// Per-(img,channel) spatial mean (stem pooling), from f32 E2.
__global__ void poolmean_k(const float* __restrict__ X, float* __restrict__ P) {
  int row = blockIdx.x;
  const float* Xi = X + (size_t)row * HW_;
  float s = 0.f;
  for (int i = threadIdx.x; i < HW_; i += 256) s += Xi[i];
  __shared__ float sm[256];
  sm[threadIdx.x] = s;
  __syncthreads();
  for (int o = 128; o > 0; o >>= 1) {
    if (threadIdx.x < o) sm[threadIdx.x] += sm[threadIdx.x + o];
    __syncthreads();
  }
  if (threadIdx.x == 0) P[row] = sm[0] * (1.0f / HW_);
}

// Stem score MLP + masked softmax fusion weights (tiny; 1 block).
__global__ void stemscore_k(const float* __restrict__ P, const float* __restrict__ w1,
                            const float* __restrict__ b1, const float* __restrict__ w2,
                            const float* __restrict__ b2,
                            const unsigned char* __restrict__ mask,
                            float* __restrict__ wts, float* __restrict__ ratio,
                            int* __restrict__ baseidx) {
  __shared__ float s[6];
  int t = threadIdx.x;
  if (t < 6) {
    const float* pr = P + t * DCH;
    float acc2 = b2[0];
    for (int h = 0; h < 24; ++h) {
      float a = b1[h];
      for (int c = 0; c < DCH; ++c) a += w1[h * DCH + c] * pr[c];
      acc2 += w2[h] * gelu_f(a);
    }
    s[t] = acc2;
  }
  __syncthreads();
  if (t < 2) {
    float m[3]; int cnt = 0;
    for (int n = 0; n < 3; ++n) { m[n] = mask[t * 3 + n] ? 1.f : 0.f; cnt += (m[n] != 0.f); }
    float lg[3], mx = -1e30f;
    for (int n = 0; n < 3; ++n) { lg[n] = (m[n] == 0.f) ? -10000.f : s[t * 3 + n]; mx = fmaxf(mx, lg[n]); }
    float e[3], sum = 0.f;
    for (int n = 0; n < 3; ++n) { e[n] = expf(lg[n] - mx); sum += e[n]; }
    float wv[3], wsum = 0.f;
    for (int n = 0; n < 3; ++n) { wv[n] = e[n] / sum * m[n]; wsum += wv[n]; }
    wsum = fmaxf(wsum, 1e-6f);
    for (int n = 0; n < 3; ++n) wts[t * 3 + n] = wv[n] / wsum;
    ratio[t] = (m[0] + m[1] + m[2]) * (1.0f / 3.0f);
    int vc = cnt < 1 ? 1 : cnt;
    baseidx[t] = vc - 1;
  }
}

// Mixed tensor -> f16 channel-last mirror [b*HW][320] (289 used, rest zero).
__global__ void fuse_mix_k(const float* __restrict__ E, const float* __restrict__ wts,
                           const int* __restrict__ baseidx, const float* __restrict__ ratio,
                           _Float16* __restrict__ MH) {
  int idx = blockIdx.x * blockDim.x + threadIdx.x;  // 2*96*HW
  if (idx >= 2 * DCH * HW_) return;
  int pix = idx % HW_;
  int t   = idx / HW_;
  int c   = t % DCH;
  int b   = t / DCH;
  const float* Eb = E + (size_t)b * 3 * DCH * HW_;
  float f = 0.f;
  for (int n = 0; n < 3; ++n) f += wts[b * 3 + n] * Eb[((size_t)n * DCH + c) * HW_ + pix];
  float ba = Eb[((size_t)baseidx[b] * DCH + c) * HW_ + pix];
  _Float16* Mb = MH + ((size_t)b * HW_ + pix) * 320;
  Mb[c]       = (_Float16)f;
  Mb[96 + c]  = (_Float16)ba;
  Mb[192 + c] = (_Float16)fabsf(f - ba);
  if (c == 0) {
    Mb[288] = (_Float16)ratio[b];
    for (int j = 289; j < 320; ++j) Mb[j] = (_Float16)0.0f;
  }
}

// L2 norms of q/k rows from f16 qkv mirror (channels 0..191 per image).
__global__ void rownorms_k(const _Float16* __restrict__ X, float* __restrict__ N) {
  int row = blockIdx.x;  // 0..383: b=row/192, c=row%192
  int b = row / 192, c = row - b * 192;
  const _Float16* p = X + ((size_t)b * 288 + c) * HW_;
  float s = 0.f;
  for (int i = threadIdx.x; i < HW_; i += 256) { float v = (float)p[i]; s += v * v; }
  __shared__ float sm[256];
  sm[threadIdx.x] = s;
  __syncthreads();
  for (int o = 128; o > 0; o >>= 1) {
    if (threadIdx.x < o) sm[threadIdx.x] += sm[threadIdx.x + o];
    __syncthreads();
  }
  if (threadIdx.x == 0) N[row] = fmaxf(sqrtf(sm[0]), 1e-12f);
}

// Raw channel Gram q.kT per (b,h): M=N=48, K=HW. One wave per 16x16 tile,
// fragments straight from f16 global (contiguous 16B/32B per lane).
__global__ void scores_wmma(const _Float16* __restrict__ QKV, float* __restrict__ Raw) {
  int bid  = blockIdx.x;          // 2*2*9
  int tile = bid % 9;
  int bh   = bid / 9;
  int b = bh >> 1, h = bh & 1;
  int ct = tile / 3, dt = tile - ct * 3;
  const _Float16* qb = QKV + ((size_t)b * 288 + h * 48 + ct * 16) * HW_;
  const _Float16* kb = QKV + ((size_t)b * 288 + 96 + h * 48 + dt * 16) * HW_;
  v8f acc = {};
  for (int k = 0; k < HW_; k += 32) {
    v16h a  = frag_a_f16(qb, HW_, k);
    v16h bb = frag_b_f16(kb, HW_, k);
    acc = wmma16(a, bb, acc);
  }
  int lane = threadIdx.x & 31;
  int n    = lane & 15;
  int moff = (lane < 16) ? 0 : 8;
  float* out = Raw + ((size_t)bh * 48 + ct * 16) * 48 + dt * 16;
#pragma unroll
  for (int r = 0; r < 8; ++r) out[(size_t)(moff + r) * 48 + n] = acc[r];
}

// scores = temp * raw/(|q||k|); softmax over d (48). One block per (b,h,c).
__global__ void attn_softmax_k(const float* __restrict__ Raw, const float* __restrict__ Nrm,
                               const float* __restrict__ temp, float* __restrict__ A) {
  int row = blockIdx.x;           // 2*2*48
  int c  = row % 48;
  int bh = row / 48;
  int b = bh >> 1, h = bh & 1;
  float nq = Nrm[b * 192 + h * 48 + c];
  float tt = temp[h];
  __shared__ float sm[64];
  int t = threadIdx.x;
  float val = 0.f, v = -1e30f;
  if (t < 48) {
    float nk = Nrm[b * 192 + 96 + h * 48 + t];
    val = tt * Raw[((size_t)bh * 48 + c) * 48 + t] / (nq * nk);
    v = val;
  }
  sm[t] = v; __syncthreads();
  for (int o = 32; o > 0; o >>= 1) { if (t < o) sm[t] = fmaxf(sm[t], sm[t + o]); __syncthreads(); }
  float mx = sm[0]; __syncthreads();
  float e = (t < 48) ? expf(val - mx) : 0.f;
  sm[t] = e; __syncthreads();
  for (int o = 32; o > 0; o >>= 1) { if (t < o) sm[t] += sm[t + o]; __syncthreads(); }
  if (t < 48) A[((size_t)bh * 48 + c) * 48 + t] = e / sm[0];
}

// o = attn @ v : M=48 ch, N=64 px (4 subtiles), K=48 pad 64. 3 waves/block.
// Output: f16 channel-last mirror [img*HW][96] (feeds proj GEMM).
__global__ void attnout_wmma(const float* __restrict__ A, const _Float16* __restrict__ QKV,
                             _Float16* __restrict__ OH) {
  __shared__ __align__(16) _Float16 lA[48 * 64];
  __shared__ __align__(16) _Float16 lV[64 * 64];
  int bh = blockIdx.y;
  int b = bh >> 1, h = bh & 1;
  int p0 = blockIdx.x * 64;
  for (int e = threadIdx.x; e < 48 * 64; e += blockDim.x) {
    int cc = e >> 6, d = e & 63;
    lA[e] = (_Float16)((d < 48) ? A[((size_t)bh * 48 + cc) * 48 + d] : 0.f);
  }
  const _Float16* vb = QKV + ((size_t)b * 288 + 192 + h * 48) * HW_ + p0;
  for (int e = threadIdx.x; e < 64 * 64; e += blockDim.x) {
    int pix = e & 63, d = e >> 6;
    lV[pix * 64 + d] = (d < 48) ? vb[(size_t)d * HW_ + pix] : (_Float16)0.0f;
  }
  __syncthreads();
  int ct = threadIdx.x >> 5;  // 0..2
  v8f acc0 = {}, acc1 = {}, acc2 = {}, acc3 = {};
#pragma unroll
  for (int kb = 0; kb < 64; kb += 32) {
    v16h a  = frag_a_f16(lA + ct * 16 * 64, 64, kb);
    v16h b0 = frag_b_f16(lV,             64, kb);
    v16h b1 = frag_b_f16(lV + 16 * 64,   64, kb);
    v16h b2 = frag_b_f16(lV + 32 * 64,   64, kb);
    v16h b3 = frag_b_f16(lV + 48 * 64,   64, kb);
    acc0 = wmma16(a, b0, acc0);
    acc1 = wmma16(a, b1, acc1);
    acc2 = wmma16(a, b2, acc2);
    acc3 = wmma16(a, b3, acc3);
  }
  int lane = threadIdx.x & 31;
  int n    = lane & 15;
  int moff = (lane < 16) ? 0 : 8;
  v8f* accs[4] = {&acc0, &acc1, &acc2, &acc3};
#pragma unroll
  for (int sub = 0; sub < 4; ++sub) {
    v8f av = *accs[sub];
    v8h o;
#pragma unroll
    for (int r = 0; r < 8; ++r) o[r] = (_Float16)av[r];
    *(v8h*)(OH + ((size_t)b * HW_ + p0 + sub * 16 + n) * DCH + h * 48 + ct * 16 + moff) = o;
  }
}

// Gated GELU: z[c] = gelu(y1[c]) * y2[c]; writes f16 mirror [b*HW][256].
__global__ void glu_k(const float* __restrict__ X, _Float16* __restrict__ YH) {
  int idx = blockIdx.x * blockDim.x + threadIdx.x;  // 2*255*HW
  if (idx >= 2 * FFH_ * HW_) return;
  int pix = idx % HW_;
  int t   = idx / HW_;
  int c   = t % FFH_;
  int b   = t / FFH_;
  const float* xb = X + (size_t)b * (2 * FFH_) * HW_;
  float y1 = xb[(size_t)c * HW_ + pix];
  float y2 = xb[(size_t)(FFH_ + c) * HW_ + pix];
  _Float16* Yb = YH + ((size_t)b * HW_ + pix) * 256;
  Yb[c] = (_Float16)(gelu_f(y1) * y2);
  if (c == 0) Yb[255] = (_Float16)0.0f;
}

// ---------------------------------------------------------------------------
// Host orchestration.
// ---------------------------------------------------------------------------
static inline int cdiv(int a, int b) { return (a + b - 1) / b; }

extern "C" void kernel_launch(void* const* d_in, const int* in_sizes, int n_in,
                              void* d_out, int out_size, void* d_ws, size_t ws_size,
                              hipStream_t stream) {
  const float* inputs = (const float*)d_in[0];
  const unsigned char* mask = (const unsigned char*)d_in[1];
  const float* se1_w = (const float*)d_in[2];  const float* se1_b = (const float*)d_in[3];
  const float* se2_w = (const float*)d_in[4];  const float* se2_b = (const float*)d_in[5];
  const float* sc1_w = (const float*)d_in[6];  const float* sc1_b = (const float*)d_in[7];
  const float* sc2_w = (const float*)d_in[8];  const float* sc2_b = (const float*)d_in[9];
  const float* mix1_w = (const float*)d_in[10]; const float* mix1_b = (const float*)d_in[11];
  const float* mix2_w = (const float*)d_in[12]; const float* mix2_b = (const float*)d_in[13];
  const float* ln1_w = (const float*)d_in[14]; const float* ln1_b = (const float*)d_in[15];
  const float* qkv_w = (const float*)d_in[16]; const float* qkv_b = (const float*)d_in[17];
  const float* dwq_w = (const float*)d_in[18]; const float* dwq_b = (const float*)d_in[19];
  const float* temp  = (const float*)d_in[20];
  const float* proj_w = (const float*)d_in[21]; const float* proj_b = (const float*)d_in[22];
  const float* ln2_w = (const float*)d_in[23]; const float* ln2_b = (const float*)d_in[24];
  const float* ffin_w = (const float*)d_in[25]; const float* ffin_b = (const float*)d_in[26];
  const float* ffdw_w = (const float*)d_in[27]; const float* ffdw_b = (const float*)d_in[28];
  const float* ffout_w = (const float*)d_in[29]; const float* ffout_b = (const float*)d_in[30];

  float* ws = (float*)d_ws;
  // f32 regions (float offsets)
  const size_t OFF_E2   = 0;                                  // 6*96*HW
  const size_t OFF_FEAT = OFF_E2 + (size_t)6 * 96 * HW_;      // 2*96*HW
  const size_t OFF_T2   = OFF_FEAT + (size_t)2 * 96 * HW_;    // 2*510*HW
  const size_t OFF_T3   = OFF_T2 + (size_t)2 * 510 * HW_;     // 2*510*HW
  const size_t OFF_SM   = OFF_T3 + (size_t)2 * 510 * HW_;     // small
  const size_t OFF_H    = OFF_SM + 32768;                     // f16 area base

  float* E2   = ws + OFF_E2;
  float* FEAT = ws + OFF_FEAT;
  float* T2   = ws + OFF_T2;
  float* T3   = ws + OFF_T3;
  float* POOL = ws + OFF_SM;            // 576
  float* WTS  = ws + OFF_SM + 576;      // 6
  float* RAT  = ws + OFF_SM + 582;      // 2
  int*   BIDX = (int*)(ws + OFF_SM + 584); // 2
  float* NRM  = ws + OFF_SM + 592;      // 384
  float* RAW  = ws + OFF_SM + 1024;     // 9216
  float* ATT  = ws + OFF_SM + 10240;    // 9216

  _Float16* hbase  = (_Float16*)(ws + OFF_H);
  _Float16* H_STEM = hbase;                               // 6*HW*96
  _Float16* H_MIX  = H_STEM + (size_t)6 * HW_ * 96;       // 2*HW*320
  _Float16* H_LN   = H_MIX  + (size_t)2 * HW_ * 320;      // 2*HW*96
  _Float16* H_GLU  = H_LN   + (size_t)2 * HW_ * 96;       // 2*HW*256
  _Float16* H_QKV  = H_GLU  + (size_t)2 * HW_ * 256;      // 2*288*HW
  _Float16* PACK   = H_QKV  + (size_t)2 * 288 * HW_;      // <=128*864

  const int TPB = 256;

  // ---- stem ----
  stem1_k<<<cdiv(6 * 96 * HW_, TPB), TPB, 0, stream>>>(inputs, se1_w, se1_b, H_STEM);
  pack_w_3x3<<<cdiv(128 * 96 * 9, TPB), TPB, 0, stream>>>(se2_w, PACK, 96, 96, 128);
  conv3x3_wmma<<<dim3(6 * HW_ / 32, 2), 128, 32 * 864 * 2, stream>>>(
      H_STEM, PACK, se2_b, E2, 96);

  poolmean_k<<<6 * 96, 256, 0, stream>>>(E2, POOL);
  stemscore_k<<<1, 64, 0, stream>>>(POOL, sc1_w, sc1_b, sc2_w, sc2_b, mask, WTS, RAT, BIDX);
  fuse_mix_k<<<cdiv(2 * 96 * HW_, TPB), TPB, 0, stream>>>(E2, WTS, BIDX, RAT, H_MIX);

  // mix1 (1x1, Cin 289 -> KPAD 320) -> gelu'd f16 mirror (feeds mix2 conv)
  pack_w_1x1<<<cdiv(128 * 320, TPB), TPB, 0, stream>>>(mix1_w, PACK, 96, 289, 320, 128);
  gemm1x1_wmma<<<dim3(2 * HW_ / 64, 2), 128, 64 * 320 * 2, stream>>>(
      H_MIX, PACK, mix1_b, nullptr, nullptr, H_LN, 320, 96, /*geluMir*/ 1);
  pack_w_3x3<<<cdiv(128 * 96 * 9, TPB), TPB, 0, stream>>>(mix2_w, PACK, 96, 96, 128);
  conv3x3_wmma<<<dim3(2 * HW_ / 32, 2), 128, 32 * 864 * 2, stream>>>(
      H_LN, PACK, mix2_b, FEAT, 96);

  // ---- transformer blocks ----
  for (int l = 0; l < LBLK; ++l) {
    // MDTA
    ln2d_k<<<cdiv(2 * HW_, TPB), TPB, 0, stream>>>(FEAT, ln1_w + l * 96, ln1_b + l * 96, H_LN, 2);
    pack_w_1x1<<<cdiv(320 * 96, TPB), TPB, 0, stream>>>(qkv_w + (size_t)l * 288 * 96, PACK, 288, 96, 96, 320);
    gemm1x1_wmma<<<dim3(2 * HW_ / 64, 5), 128, 64 * 96 * 2, stream>>>(
        H_LN, PACK, qkv_b + l * 288, nullptr, T2, nullptr, 96, 288, 0);
    dwconv3x3_k<<<cdiv(2 * 288 * HW_, TPB), TPB, 0, stream>>>(
        T2, dwq_w + (size_t)l * 288 * 9, dwq_b + l * 288, nullptr, H_QKV, 288, 2);
    rownorms_k<<<384, 256, 0, stream>>>(H_QKV, NRM);
    scores_wmma<<<2 * 2 * 9, 32, 0, stream>>>(H_QKV, RAW);
    attn_softmax_k<<<2 * 2 * 48, 64, 0, stream>>>(RAW, NRM, temp + l * HEADS, ATT);
    attnout_wmma<<<dim3(HW_ / 64, 4), 96, 0, stream>>>(ATT, H_QKV, H_LN);
    pack_w_1x1<<<cdiv(128 * 96, TPB), TPB, 0, stream>>>(proj_w + (size_t)l * 96 * 96, PACK, 96, 96, 96, 128);
    gemm1x1_wmma<<<dim3(2 * HW_ / 64, 2), 128, 64 * 96 * 2, stream>>>(
        H_LN, PACK, proj_b + l * 96, /*resid*/ FEAT, FEAT, nullptr, 96, 96, 0);

    // gated-dconv FFN
    ln2d_k<<<cdiv(2 * HW_, TPB), TPB, 0, stream>>>(FEAT, ln2_w + l * 96, ln2_b + l * 96, H_LN, 2);
    pack_w_1x1<<<cdiv(512 * 96, TPB), TPB, 0, stream>>>(ffin_w + (size_t)l * 510 * 96, PACK, 510, 96, 96, 512);
    gemm1x1_wmma<<<dim3(2 * HW_ / 64, 8), 128, 64 * 96 * 2, stream>>>(
        H_LN, PACK, ffin_b + l * 510, nullptr, T2, nullptr, 96, 510, 0);
    dwconv3x3_k<<<cdiv(2 * 510 * HW_, TPB), TPB, 0, stream>>>(
        T2, ffdw_w + (size_t)l * 510 * 9, ffdw_b + l * 510, T3, nullptr, 510, 2);
    glu_k<<<cdiv(2 * 255 * HW_, TPB), TPB, 0, stream>>>(T3, H_GLU);
    pack_w_1x1<<<cdiv(128 * 256, TPB), TPB, 0, stream>>>(ffout_w + (size_t)l * 96 * 255, PACK, 96, 255, 256, 128);
    gemm1x1_wmma<<<dim3(2 * HW_ / 64, 2), 128, 64 * 256 * 2, stream>>>(
        H_GLU, PACK, ffout_b + l * 96, /*resid*/ FEAT, FEAT, nullptr, 256, 96, 0);
  }

  hipMemcpyAsync(d_out, FEAT, (size_t)2 * 96 * HW_ * sizeof(float),
                 hipMemcpyDeviceToDevice, stream);
}
